// LineVectorizer_55929064129108
// MI455X (gfx1250) — matrix operand order repaired and checked by reading, not compiled, not run on hardware
//
#include <hip/hip_runtime.h>
#include <hip/hip_bf16.h>

// ---------------------------------------------------------------------------
// LineVectorizer head for MI455X (gfx1250, wave32, WMMA 16x16x32 f16).
//   K1: fc1 1x1-conv GEMM  -> x[b][pix][128]   (pixel-major for coalesced gathers)
//   K2: bilinear line sampling + MaxPool1d(4)  -> xp[n][128][8]
//   K3: Bottleneck1D (conv1/conv2(k3)/conv3 all via WMMA) + residual + fc2
// ---------------------------------------------------------------------------

typedef __attribute__((ext_vector_type(16))) _Float16 v16h;
typedef __attribute__((ext_vector_type(8)))  _Float16 v8h;
typedef __attribute__((ext_vector_type(8)))  float    v8f;

__device__ __forceinline__ v8f wmma_f16(v16h a, v16h b, v8f c) {
  // D = A(16x32 f16) * B(32x16 f16) + C(16x16 f32)
  return __builtin_amdgcn_wmma_f32_16x16x32_f16(
      /*neg_a=*/false, a, /*neg_b=*/false, b,
      /*c_mod=*/(short)0, c, /*reuse_a=*/false, /*reuse_b=*/false);
}

// A-operand k order chosen here (contiguous, B frags match; per lane, hi = lane>=16):
//   halves 0..7  -> K = kk + hi*16 + i
//   halves 8..15 -> K = kk + hi*16 + 8 + i
__device__ __forceinline__ v16h load_a_lds(const _Float16* W, int rs, int m,
                                           int kk, int hi8) {
  const _Float16* r0 = W + m * rs + kk + 2 * hi8;
  v8h x0 = *(const v8h*)r0;
  v8h x1 = *(const v8h*)(r0 + 8);
  return __builtin_shufflevector(x0, x1, 0,1,2,3,4,5,6,7,8,9,10,11,12,13,14,15);
}

__device__ __forceinline__ v16h load_a_global_f32(const float* W, int rs, int m,
                                                  int kk, int hi8) {
  const float* r0 = W + m * rs + kk + 2 * hi8;
  v16h a;
#pragma unroll
  for (int i = 0; i < 8; ++i) {
    a[i]     = (_Float16)r0[i];
    a[i + 8] = (_Float16)r0[8 + i];
  }
  return a;
}

// ---------------------------------------------------------------------------
// Kernel 1: x[b][pix][m] = sum_k fc1_w[m][k] * feature[b][k][pix] + fc1_b[m]
// Block: 256 thr (8 waves). Tile: M=128 (wave m-rows), N=128 pixels, K-step 32.
// ---------------------------------------------------------------------------
__global__ __launch_bounds__(256) void k1_fc1_gemm(
    const float* __restrict__ feature, const float* __restrict__ fc1_w,
    const float* __restrict__ fc1_b, float* __restrict__ x) {
  __shared__ _Float16 sW[128 * 32];   // [m][k]   (8 KB)
  __shared__ _Float16 sF[128 * 48];   // [p][k] pad-48 for 32B-aligned b-frags

  const int tid = threadIdx.x;
  const int lane = tid & 31, wid = tid >> 5;
  const int lane15 = lane & 15, hi = lane >> 4, hi8 = hi * 8;
  const int bid = blockIdx.x;
  const int b = bid >> 9;                 // 512 pixel-tiles per image
  const int pix0 = (bid & 511) * 128;

  const v8f vzero = {0.f,0.f,0.f,0.f,0.f,0.f,0.f,0.f};
  v8f acc[8];
#pragma unroll
  for (int nt = 0; nt < 8; ++nt) acc[nt] = vzero;

  for (int kk = 0; kk < 256; kk += 32) {
    __syncthreads();
    // stage weight slice [128 m][32 k]
#pragma unroll
    for (int i = 0; i < 16; ++i) {
      int e = i * 256 + tid;
      int m = e >> 5, k = e & 31;
      sW[m * 32 + k] = (_Float16)fc1_w[m * 256 + kk + k];
    }
    // stage feature slice transposed: [128 p][32 k] (coalesced global rows)
#pragma unroll
    for (int i = 0; i < 16; ++i) {
      int e = i * 256 + tid;
      int k = e >> 7, p = e & 127;
      sF[p * 48 + k] =
          (_Float16)feature[(size_t)(b * 256 + kk + k) * 65536 + pix0 + p];
    }
    __syncthreads();

    const v16h afrag = load_a_lds(sW, 32, wid * 16 + lane15, 0, hi8);
#pragma unroll
    for (int nt = 0; nt < 8; ++nt) {
      const int p = nt * 16 + lane15;
      v16h bfrag = *(const v16h*)&sF[p * 48 + hi * 16];  // K = hi*16 + h
      acc[nt] = wmma_f16(afrag, bfrag, acc[nt]);
    }
  }

  // store pixel-major: x[(b*65536+n)*128 + m]
#pragma unroll
  for (int nt = 0; nt < 8; ++nt) {
    const int n = pix0 + nt * 16 + lane15;
    float* xb = x + ((size_t)b * 65536 + n) * 128;
#pragma unroll
    for (int r = 0; r < 8; ++r) {
      const int m = wid * 16 + r + hi8;
      xb[m] = acc[nt][r] + fc1_b[m];
    }
  }
}

// ---------------------------------------------------------------------------
// Kernel 2: bilinear sample 32 pts per line over x[b][pix][c], MaxPool1d(4)
// Block = 1 line, 128 threads = channels. Corner gathers are contiguous 512B.
// ---------------------------------------------------------------------------
__global__ __launch_bounds__(128) void k2_sample_pool(
    const float* __restrict__ lines, const float* __restrict__ x,
    float* __restrict__ xp) {
  __shared__ int   sp00[32], sp10[32], sp01[32], sp11[32];
  __shared__ float sw00[32], sw10[32], sw01[32], sw11[32];

  const int n = blockIdx.x;        // 0..9999
  const int b = n / 5000;
  const int tid = threadIdx.x;

  if (tid < 32) {
    const float* lp = lines + (size_t)n * 4;   // [e0.x, e0.y, e1.x, e1.y]
    const float lam = tid * (1.0f / 31.0f);
    const float px = lp[0] * lam + lp[2] * (1.0f - lam) - 0.5f;
    const float py = lp[1] * lam + lp[3] * (1.0f - lam) - 0.5f;
    const float px0 = fminf(fmaxf(floorf(px), 0.f), 255.f);
    const float py0 = fminf(fmaxf(floorf(py), 0.f), 255.f);
    const float px1 = fminf(px0 + 1.f, 255.f);
    const float py1 = fminf(py0 + 1.f, 255.f);
    const int ix0 = (int)px0, iy0 = (int)py0, ix1 = (int)px1, iy1 = (int)py1;
    sp00[tid] = ix0 * 256 + iy0;  sp10[tid] = ix1 * 256 + iy0;
    sp01[tid] = ix0 * 256 + iy1;  sp11[tid] = ix1 * 256 + iy1;
    sw00[tid] = (px1 - px) * (py1 - py);
    sw10[tid] = (px - px0) * (py1 - py);
    sw01[tid] = (px1 - px) * (py - py0);
    sw11[tid] = (px - px0) * (py - py0);
  }
  __syncthreads();

  const int c = tid;  // channel
  const float* xb = x + (size_t)b * 65536 * 128;
  float* xpn = xp + (size_t)n * 1024 + c * 8;
#pragma unroll
  for (int t = 0; t < 8; ++t) {
    float mx = -3.402823e38f;
#pragma unroll
    for (int q = 0; q < 4; ++q) {
      const int p = t * 4 + q;
      const float v = sw00[p] * xb[(size_t)sp00[p] * 128 + c] +
                      sw10[p] * xb[(size_t)sp10[p] * 128 + c] +
                      sw01[p] * xb[(size_t)sp01[p] * 128 + c] +
                      sw11[p] * xb[(size_t)sp11[p] * 128 + c];
      mx = fmaxf(mx, v);
    }
    xpn[t] = mx;
  }
}

// ---------------------------------------------------------------------------
// Kernel 3: Bottleneck1D + residual + fc2. 128 thr = 4 waves; each wave owns
// 2 lines -> WMMA N-tile of 16 columns (col = line*8 + t).
// ---------------------------------------------------------------------------
__global__ __launch_bounds__(128) void k3_head(
    const float* __restrict__ xp,
    const float* __restrict__ conv1_w, const float* __restrict__ conv1_b,
    const float* __restrict__ bn1_g, const float* __restrict__ bn1_b,
    const float* __restrict__ bn1_m, const float* __restrict__ bn1_v,
    const float* __restrict__ bn2_g, const float* __restrict__ bn2_b,
    const float* __restrict__ bn2_m, const float* __restrict__ bn2_v,
    const float* __restrict__ conv2_w, const float* __restrict__ conv2_b,
    const float* __restrict__ bn3_g, const float* __restrict__ bn3_b,
    const float* __restrict__ bn3_m, const float* __restrict__ bn3_v,
    const float* __restrict__ conv3_w, const float* __restrict__ conv3_b,
    const float* __restrict__ fc2_w, const float* __restrict__ fc2_b,
    float* __restrict__ out) {
  __shared__ _Float16 sW2[3 * 64 * 64];  // [dt][m][ci]  24 KB
  __shared__ _Float16 sW3[128 * 64];     // [m][ci]      16 KB
  __shared__ _Float16 sBuf[4][2048];     // per-wave reuse: h1 / h2pad / h3
  __shared__ float sB1i[128], sB1a[128], sB2i[64], sB2a[64], sB3i[64], sB3a[64];

  const int tid = threadIdx.x;
  {
    const int c = tid;  // 0..127
    float i1 = bn1_g[c] * rsqrtf(bn1_v[c] + 1e-5f);
    sB1i[c] = i1;  sB1a[c] = bn1_b[c] - bn1_m[c] * i1;
    if (c < 64) {
      float i2 = bn2_g[c] * rsqrtf(bn2_v[c] + 1e-5f);
      sB2i[c] = i2;  sB2a[c] = bn2_b[c] - bn2_m[c] * i2;
      float i3 = bn3_g[c] * rsqrtf(bn3_v[c] + 1e-5f);
      sB3i[c] = i3;  sB3a[c] = bn3_b[c] - bn3_m[c] * i3;
    }
  }
  for (int e = tid; e < 12288; e += 128) {   // conv2_w [m][ci][dt] -> [dt][m][ci]
    const int dt = e >> 12, rem = e & 4095, m = rem >> 6, ci = rem & 63;
    sW2[e] = (_Float16)conv2_w[(m * 64 + ci) * 3 + dt];
  }
  for (int e = tid; e < 8192; e += 128) sW3[e] = (_Float16)conv3_w[e];
  __syncthreads();

  const int lane = tid & 31, w = tid >> 5;
  const int lane15 = lane & 15, hi = lane >> 4, hi8 = hi * 8;
  const int col = lane15, line = col >> 3, t = col & 7;
  const int n0 = blockIdx.x * 8 + w * 2;
  _Float16* buf = sBuf[w];
  const v8f vzero = {0.f,0.f,0.f,0.f,0.f,0.f,0.f,0.f};

  // ---- h1 = relu(bn1(xp)) staged as [col(16)][c(128)] f16 ----
  for (int e = lane; e < 2048; e += 32) {
    const int cE = e >> 7, c = e & 127, lE = cE >> 3, tE = cE & 7;
    const float v = xp[(size_t)(n0 + lE) * 1024 + c * 8 + tE];
    buf[cE * 128 + c] = (_Float16)fmaxf(v * sB1i[c] + sB1a[c], 0.f);
  }
  __syncthreads();

  // ---- GEMM1: out1[64,16] = W1[64,128] x h1[128,16] ----
  v8f acc1[4];
#pragma unroll
  for (int mt = 0; mt < 4; ++mt) acc1[mt] = vzero;
#pragma unroll
  for (int kkI = 0; kkI < 4; ++kkI) {
    const int kk = kkI * 32;
    v16h bf = *(const v16h*)&buf[col * 128 + kk + hi * 16];
#pragma unroll
    for (int mt = 0; mt < 4; ++mt) {
      v16h af = load_a_global_f32(conv1_w, 128, mt * 16 + lane15, kk, hi8);
      acc1[mt] = wmma_f16(af, bf, acc1[mt]);
    }
  }
  __syncthreads();

  // ---- h2 = relu(bn2(out1+b1)), stored zero-padded [pcol(20)][ci(64)] ----
  for (int e = lane; e < 1280; e += 32) buf[e] = (_Float16)0.f;
#pragma unroll
  for (int mt = 0; mt < 4; ++mt)
#pragma unroll
    for (int r = 0; r < 8; ++r) {
      const int m = mt * 16 + r + hi8;
      const float v = acc1[mt][r] + conv1_b[m];
      buf[(line * 10 + t + 1) * 64 + m] =
          (_Float16)fmaxf(v * sB2i[m] + sB2a[m], 0.f);
    }
  __syncthreads();

  // ---- conv2 (k=3, pad 1) as 3 shifted WMMA GEMMs ----
  v8f acc2[4];
#pragma unroll
  for (int mt = 0; mt < 4; ++mt) acc2[mt] = vzero;
#pragma unroll
  for (int dt = 0; dt < 3; ++dt) {
    const int pcol = line * 10 + t + dt;  // = padded(t + dt - 1)
#pragma unroll
    for (int kkI = 0; kkI < 2; ++kkI) {
      const int kk = kkI * 32;
      v16h bf = *(const v16h*)&buf[pcol * 64 + kk + hi * 16];
#pragma unroll
      for (int mt = 0; mt < 4; ++mt) {
        v16h af = load_a_lds(sW2 + dt * 4096, 64, mt * 16 + lane15, kk, hi8);
        acc2[mt] = wmma_f16(af, bf, acc2[mt]);
      }
    }
  }
  __syncthreads();

  // ---- h3 = relu(bn3(out2+b2)) staged [col(16)][ci(64)] ----
#pragma unroll
  for (int mt = 0; mt < 4; ++mt)
#pragma unroll
    for (int r = 0; r < 8; ++r) {
      const int m = mt * 16 + r + hi8;
      const float v = acc2[mt][r] + conv2_b[m];
      buf[col * 64 + m] = (_Float16)fmaxf(v * sB3i[m] + sB3a[m], 0.f);
    }
  __syncthreads();

  // ---- GEMM3: out3[128,16] = W3[128,64] x h3[64,16] ----
  v8f acc3[8];
#pragma unroll
  for (int mt = 0; mt < 8; ++mt) acc3[mt] = vzero;
#pragma unroll
  for (int kkI = 0; kkI < 2; ++kkI) {
    const int kk = kkI * 32;
    v16h bf = *(const v16h*)&buf[col * 64 + kk + hi * 16];
#pragma unroll
    for (int mt = 0; mt < 8; ++mt) {
      v16h af = load_a_lds(sW3, 64, mt * 16 + lane15, kk, hi8);
      acc3[mt] = wmma_f16(af, bf, acc3[mt]);
    }
  }

  // ---- residual + relu + fc2 partial dot, wave32 xor-reduction ----
  float p0 = 0.f, p1 = 0.f, p2 = 0.f;
  const float* xpl = xp + (size_t)(n0 + line) * 1024;
#pragma unroll
  for (int mt = 0; mt < 8; ++mt)
#pragma unroll
    for (int r = 0; r < 8; ++r) {
      const int m = mt * 16 + r + hi8;
      const float y = acc3[mt][r] + conv3_b[m] + xpl[m * 8 + t];
      const float f = fmaxf(y, 0.f);
      const int fi = m * 8 + t;
      p0 += f * fc2_w[fi];
      p1 += f * fc2_w[1024 + fi];
      p2 += f * fc2_w[2048 + fi];
    }
  // reduce over lanes {hi}x{t} of this line: xor 16,4,2,1 (bit3 = line kept)
  p0 += __shfl_xor(p0, 16); p0 += __shfl_xor(p0, 4);
  p0 += __shfl_xor(p0, 2);  p0 += __shfl_xor(p0, 1);
  p1 += __shfl_xor(p1, 16); p1 += __shfl_xor(p1, 4);
  p1 += __shfl_xor(p1, 2);  p1 += __shfl_xor(p1, 1);
  p2 += __shfl_xor(p2, 16); p2 += __shfl_xor(p2, 4);
  p2 += __shfl_xor(p2, 2);  p2 += __shfl_xor(p2, 1);
  if (hi == 0 && t == 0) {
    float* o = out + (size_t)(n0 + line) * 3;
    o[0] = p0 + fc2_b[0];
    o[1] = p1 + fc2_b[1];
    o[2] = p2 + fc2_b[2];
  }
}

// ---------------------------------------------------------------------------
extern "C" void kernel_launch(void* const* d_in, const int* in_sizes, int n_in,
                              void* d_out, int out_size, void* d_ws,
                              size_t ws_size, hipStream_t stream) {
  const float* feature = (const float*)d_in[0];
  const float* lines   = (const float*)d_in[1];
  const float* fc1_w   = (const float*)d_in[2];
  const float* fc1_b   = (const float*)d_in[3];
  const float* bn1_g   = (const float*)d_in[4];
  const float* bn1_b   = (const float*)d_in[5];
  const float* bn1_m   = (const float*)d_in[6];
  const float* bn1_v   = (const float*)d_in[7];
  const float* conv1_w = (const float*)d_in[8];
  const float* conv1_b = (const float*)d_in[9];
  const float* bn2_g   = (const float*)d_in[10];
  const float* bn2_b   = (const float*)d_in[11];
  const float* bn2_m   = (const float*)d_in[12];
  const float* bn2_v   = (const float*)d_in[13];
  const float* conv2_w = (const float*)d_in[14];
  const float* conv2_b = (const float*)d_in[15];
  const float* bn3_g   = (const float*)d_in[16];
  const float* bn3_b   = (const float*)d_in[17];
  const float* bn3_m   = (const float*)d_in[18];
  const float* bn3_v   = (const float*)d_in[19];
  const float* conv3_w = (const float*)d_in[20];
  const float* conv3_b = (const float*)d_in[21];
  const float* fc2_w   = (const float*)d_in[22];
  const float* fc2_b   = (const float*)d_in[23];
  float* out = (float*)d_out;

  float* ws = (float*)d_ws;
  float* x  = ws;                       // [2][65536][128] = 16,777,216 floats
  float* xp = ws + 16777216;            // [10000][128][8] = 10,240,000 floats

  k1_fc1_gemm<<<1024, 256, 0, stream>>>(feature, fc1_w, fc1_b, x);
  k2_sample_pool<<<10000, 128, 0, stream>>>(lines, x, xp);
  k3_head<<<1250, 128, 0, stream>>>(xp, conv1_w, conv1_b, bn1_g, bn1_b, bn1_m,
                                    bn1_v, bn2_g, bn2_b, bn2_m, bn2_v, conv2_w,
                                    conv2_b, bn3_g, bn3_b, bn3_m, bn3_v,
                                    conv3_w, conv3_b, fc2_w, fc2_b, out);
}
